// PureAEVComputer_80032420593980
// MI455X (gfx1250) — compile-verified, matrix operand order checked by reading, not run on hardware
//
#include <hip/hip_runtime.h>
#include <hip/hip_bf16.h>

typedef __attribute__((ext_vector_type(16))) _Float16 v16h;
typedef __attribute__((ext_vector_type(8)))  float    v8f;

#define RCR_F        5.2f
#define PI_OVER_RCR  (3.14159265358979f / 5.2f)
#define ETA_LOG2E    23.083120654223414f   // 16 * log2(e): exp(-16 t^2) = exp2(-ETA_LOG2E t^2)
#define NBATCH       4
#define NATOMS       1024
#define OUTW         1008
#define WPB          2                     // waves per block (wave32)
#define CHUNKS       (NATOMS / 32)         // 32 neighbor chunks of 32

// K-index (neighbor j within a 32-chunk) for half-slot h of `lane`,
// per CDNA5 ISA 16-bit A/B 16x32 VGPR layout (CDNA5 ISA 7.12.2).
__device__ __forceinline__ int jmap(int lane, int h) {
    const int v = h >> 1;
    const int base = (v < 4) ? (v * 2) : (v * 2 + 8);
    return base + (h & 1) + ((lane & 16) ? 8 : 0);
}

// Wave-local LDS ordering: LDS ops from one wave complete in order (DScnt),
// we only need to stop compiler reordering across the producer/consumer point.
__device__ __forceinline__ void wave_lds_fence() {
    __builtin_amdgcn_fence(__ATOMIC_RELEASE, "wavefront");
    __builtin_amdgcn_wave_barrier();
    __builtin_amdgcn_fence(__ATOMIC_ACQUIRE, "wavefront");
}

__global__ void __launch_bounds__(WPB * 32)
aev_radial_kernel(const int* __restrict__ species,
                  const float* __restrict__ coords,
                  float* __restrict__ out)
{
    __shared__ float xs[NATOMS], ys[NATOMS], zs[NATOMS];
    __shared__ alignas(32) _Float16 Bl[CHUNKS][32][16];  // one-hot B tiles, WMMA reg layout
    __shared__ float d_lds[WPB][NATOMS];
    __shared__ float fc_lds[WPB][NATOMS];

    const int tid   = threadIdx.x;
    const int lane  = tid & 31;
    const int w     = tid >> 5;
    const int gwave = blockIdx.x * WPB + w;      // tile id: 16 atoms per wave
    const int batch = gwave >> 6;                // 64 tiles per batch
    const int i_base = (gwave & 63) * 16;

    const int cbase = batch * NATOMS * 3;
    const int sbase = batch * NATOMS;
    __builtin_prefetch(&coords[cbase], 0, 0);
    __builtin_prefetch(&species[sbase], 0, 0);

    // ---- stage batch coords into LDS (SoA) ----
    for (int j = tid; j < NATOMS; j += WPB * 32) {
        xs[j] = coords[cbase + 3 * j + 0];
        ys[j] = coords[cbase + 3 * j + 1];
        zs[j] = coords[cbase + 3 * j + 2];
    }
    // ---- precompute one-hot B tiles (species only, shared by both waves) ----
    for (int idx = tid; idx < CHUNKS * 32; idx += WPB * 32) {
        const int c = idx >> 5;
        const int L = idx & 31;
        const int n = L & 15;                    // WMMA column = species slot
        #pragma unroll
        for (int h = 0; h < 16; ++h) {
            const int j = c * 32 + jmap(L, h);
            const int sp = species[sbase + j];
            Bl[c][L][h] = (n < 4 && sp == n) ? (_Float16)1.0f : (_Float16)0.0f;
        }
    }
    __syncthreads();   // cross-wave: coords + B tiles ready

    const int   k        = lane & 15;            // WMMA row = shell index
    const float shf      = 0.9f + 0.26875f * (float)k;
    const int   half_off = (lane & 16) ? 8 : 0;  // C-row offset for upper half-wave

    for (int isub = 0; isub < 16; ++isub) {
        const int   i  = i_base + isub;
        const float xi = xs[i], yi = ys[i], zi = zs[i];

        // ---- distance + cutoff pass: one sqrt/cos per pair, stored in LDS ----
        for (int j = lane; j < NATOMS; j += 32) {
            const float dx = xi - xs[j], dy = yi - ys[j], dz = zi - zs[j];
            const float d2 = dx * dx + dy * dy + dz * dz;
            const float d  = (d2 > 0.0f) ? sqrtf(d2) : 0.0f;
            float fc = 0.0f;
            if (d > 0.0f && d <= RCR_F)
                fc = 0.5f * __cosf(PI_OVER_RCR * d) + 0.5f;
            d_lds[w][j]  = d;
            fc_lds[w][j] = fc;
        }
        wave_lds_fence();  // d/fc are wave-private; LDS is in-order per wave

        // ---- C[k,s] += sum_j fc*exp2(-eta*log2e*(d-shf_k)^2) * onehot[j,s] ----
        v8f acc = {};
        #pragma unroll 2
        for (int c = 0; c < CHUNKS; ++c) {
            const int j0 = c * 32;
            union { v16h v; _Float16 h[16]; } a;
            #pragma unroll
            for (int h = 0; h < 16; ++h) {
                const int   jl = jmap(lane, h);
                const float d  = d_lds[w][j0 + jl];
                const float fc = fc_lds[w][j0 + jl];
                const float t  = d - shf;
                const float g  = fc * __builtin_amdgcn_exp2f(-ETA_LOG2E * t * t);
                a.h[h] = (_Float16)g;
            }
            const v16h b = *(const v16h*)Bl[c][lane];
            acc = __builtin_amdgcn_wmma_f32_16x16x32_f16(
                false, a.v, false, b, (short)0, acc, false, false);
        }

        // ---- store C: lane column = species s (<4), rows k = half_off + r ----
        const int s = lane & 15;
        if (s < 4) {
            float* o = out + (size_t)(batch * NATOMS + i) * OUTW + s * 16 + half_off;
            union { v8f v; float4 f4[2]; } u; u.v = acc;
            *(float4*)(o)     = u.f4[0];
            *(float4*)(o + 4) = u.f4[1];
        }
        wave_lds_fence();  // before next atom overwrites d_lds/fc_lds
    }
}

__global__ void fill_zero_kernel(float4* __restrict__ out4, int n4) {
    const int stride = gridDim.x * blockDim.x;
    for (int i = blockIdx.x * blockDim.x + threadIdx.x; i < n4; i += stride)
        out4[i] = make_float4(0.f, 0.f, 0.f, 0.f);
}

extern "C" void kernel_launch(void* const* d_in, const int* in_sizes, int n_in,
                              void* d_out, int out_size, void* d_ws, size_t ws_size,
                              hipStream_t stream) {
    const int*   species = (const int*)d_in[0];
    const float* coords  = (const float*)d_in[1];
    float*       out     = (float*)d_out;

    // 1) zero whole output (covers the 944-wide pad; harness poisons d_out)
    const int n4 = out_size / 4;  // out_size = 4*1024*1008, divisible by 4
    fill_zero_kernel<<<1024, 256, 0, stream>>>((float4*)out, n4);

    // 2) radial AEV: 256 wave-tiles (16 atoms each), 2 waves per block
    const int tiles  = NBATCH * (NATOMS / 16);  // 256
    const int blocks = tiles / WPB;             // 128
    aev_radial_kernel<<<blocks, WPB * 32, 0, stream>>>(species, coords, out);
}
